// QuantizedLinear_77756087927527
// MI455X (gfx1250) — compile-verified, matrix operand order checked by reading, not run on hardware
//
#include <hip/hip_runtime.h>
#include <hip/hip_bf16.h>

typedef __attribute__((ext_vector_type(16))) _Float16   v16h;
typedef __attribute__((ext_vector_type(2)))  __fp16     v2fp16;
typedef __attribute__((ext_vector_type(8)))  float      v8f;
typedef __attribute__((ext_vector_type(4)))  unsigned int u32x4;

#define BK  32
#define LDT 40   // LDS row stride in f16 elems: 32 data + 8 pad = 80B -> conflict-free b128 reads

// fast-path block tile
#define GBM 128
#define GBN 256
// fallback block tile
#define FBM 128
#define FBN 128

// ---------- helpers ----------

__device__ __forceinline__ unsigned pk2(float a, float b) {
    v2fp16 h = __builtin_amdgcn_cvt_pkrtz(a, b);  // v_cvt_pk_rtz_f16_f32
    union { v2fp16 h; unsigned u; } c; c.h = h; return c.u;
}

// ternary quantize: |w| < th -> 0 else sign(w)
__device__ __forceinline__ float qtern(float w, float th) {
    float s = __builtin_copysignf(1.0f, w);
    return (__builtin_fabsf(w) >= th) ? s : 0.0f;
}

// Load one 16x32 f16 fragment from an LDS tile (rows contiguous in K, stride LDT).
// CDNA5 16-bit A/B layout:
//   lanes 0-15 : VGPR0-3 = K 0..7,  VGPR4-7 = K 16..23   (row = lane)
//   lanes 16-31: VGPR0-3 = K 8..15, VGPR4-7 = K 24..31   (row = lane-16)
__device__ __forceinline__ v16h load_frag(const _Float16* tile, int lane) {
    const int r  = lane & 15;
    const int kh = (lane & 16) ? 8 : 0;
    const _Float16* p = tile + r * LDT + kh;
    v16h f;
    ((u32x4*)&f)[0] = *(const u32x4*)(p);        // ds_load_b128
    ((u32x4*)&f)[1] = *(const u32x4*)(p + 16);   // ds_load_b128
    return f;
}

// ---------- pre-pass kernels (one-time, bandwidth bound) ----------

__global__ __launch_bounds__(256)
void pack_x_f16(const float* __restrict__ X, u32x4* __restrict__ Xh, long n8) {
    const long i = (long)blockIdx.x * 256 + threadIdx.x;   // handles 8 floats
    if (i >= n8) return;
    const float4* s = (const float4*)X;
    float4 a = s[2 * i], b = s[2 * i + 1];
    u32x4 o;
    o[0] = pk2(a.x, a.y); o[1] = pk2(a.z, a.w);
    o[2] = pk2(b.x, b.y); o[3] = pk2(b.z, b.w);
    Xh[i] = o;
}

__global__ __launch_bounds__(256)
void quant_w_f16(const float* __restrict__ W, u32x4* __restrict__ Qh,
                 const float* __restrict__ alpha, long n8) {
    const long i = (long)blockIdx.x * 256 + threadIdx.x;
    if (i >= n8) return;
    const float th = 0.5f * (__builtin_fabsf(alpha[0]) + 1e-8f);
    const float4* s = (const float4*)W;
    float4 a = s[2 * i], b = s[2 * i + 1];
    u32x4 o;
    o[0] = pk2(qtern(a.x, th), qtern(a.y, th));
    o[1] = pk2(qtern(a.z, th), qtern(a.w, th));
    o[2] = pk2(qtern(b.x, th), qtern(b.y, th));
    o[3] = pk2(qtern(b.z, th), qtern(b.w, th));
    Qh[i] = o;
}

// ---------- fast path: 128x256 block, 8 waves of 64x64, 16 WMMA/iter/wave ----------

__global__ __launch_bounds__(256)
void gemm_f16_wmma(const _Float16* __restrict__ Xh,   // [M,K] f16
                   const _Float16* __restrict__ Qh,   // [N,K] f16 ternary
                   const float* __restrict__ alpha,
                   const float* __restrict__ bias,
                   float* __restrict__ Y,             // [M,N] f32
                   int M, int N, int K)
{
    __shared__ __align__(16) _Float16 As[2][GBM * LDT];   // 20 KB
    __shared__ __align__(16) _Float16 Bs[2][GBN * LDT];   // 40 KB

    const int tid   = threadIdx.x;
    const int lane  = tid & 31;
    const int wave  = tid >> 5;
    const int waveM = wave & 1;    // 2 waves along M, 64 rows each
    const int waveN = wave >> 1;   // 4 waves along N, 64 cols each

    const int blockM = blockIdx.y * GBM;
    const int blockN = blockIdx.x * GBN;

    const float alpha_eff = __builtin_fabsf(alpha[0]) + 1e-8f;

    // staging: thread t copies 16B at (row = t/4 + 64*p, halves (t%4)*8)
    const int lr = tid >> 2;          // 0..63
    const int lq = (tid & 3) << 3;    // 0,8,16,24 halves
    const _Float16* xa = Xh + (size_t)(blockM + lr) * K + lq;
    const _Float16* qa = Qh + (size_t)(blockN + lr) * K + lq;

    u32x4 aReg[2], bReg[4];
    const int KT = K / BK;

    auto loadRegs = [&](int kt) {
        const int k0 = kt * BK;
#pragma unroll
        for (int p = 0; p < 2; ++p)
            aReg[p] = *(const u32x4*)(xa + (size_t)(64 * p) * K + k0);
#pragma unroll
        for (int p = 0; p < 4; ++p)
            bReg[p] = *(const u32x4*)(qa + (size_t)(64 * p) * K + k0);
        if (kt + 1 < KT) {  // global_prefetch_b8 of the tile after next
            __builtin_prefetch(xa + k0 + BK, 0, 1);
            __builtin_prefetch(qa + k0 + BK, 0, 1);
        }
    };

    auto storeLDS = [&](int st) {
#pragma unroll
        for (int p = 0; p < 2; ++p)
            *(u32x4*)&As[st][(lr + 64 * p) * LDT + lq] = aReg[p];   // 16B aligned
#pragma unroll
        for (int p = 0; p < 4; ++p)
            *(u32x4*)&Bs[st][(lr + 64 * p) * LDT + lq] = bReg[p];
    };

    v8f acc[4][4];
#pragma unroll
    for (int i = 0; i < 4; ++i)
#pragma unroll
        for (int j = 0; j < 4; ++j)
            acc[i][j] = (v8f)(0.0f);

    loadRegs(0);
    storeLDS(0);
    __syncthreads();

    for (int kt = 0; kt < KT; ++kt) {
        const int cur = kt & 1;
        const bool haveNext = (kt + 1) < KT;

        if (haveNext) loadRegs(kt + 1);     // VMEM overlaps WMMA below

        const _Float16* Ab = &As[cur][(waveM * 64) * LDT];
        const _Float16* Bb = &Bs[cur][(waveN * 64) * LDT];

        v16h af[4];
#pragma unroll
        for (int i = 0; i < 4; ++i)
            af[i] = load_frag(Ab + i * 16 * LDT, lane);

#pragma unroll
        for (int j = 0; j < 4; ++j) {
            v16h bf = load_frag(Bb + j * 16 * LDT, lane);
#pragma unroll
            for (int i = 0; i < 4; ++i)
                acc[i][j] = __builtin_amdgcn_wmma_f32_16x16x32_f16(
                    false, af[i], false, bf, (short)0, acc[i][j], false, false);
        }

        if (haveNext) storeLDS((kt + 1) & 1);
        __syncthreads();
    }

#pragma unroll
    for (int i = 0; i < 4; ++i) {
        const int rbase = blockM + waveM * 64 + i * 16 + ((lane & 16) ? 8 : 0);
#pragma unroll
        for (int j = 0; j < 4; ++j) {
            const int col = blockN + waveN * 64 + j * 16 + (lane & 15);
            const float bv = bias[col];
#pragma unroll
            for (int v = 0; v < 8; ++v)
                Y[(size_t)(rbase + v) * N + col] = alpha_eff * acc[i][j][v] + bv;
        }
    }
}

// ---------- fallback: fused convert+quantize in-kernel (if ws too small) ----------

__device__ __forceinline__ unsigned long long pack4h(float a, float b, float c, float d) {
    union { unsigned u[2]; unsigned long long q; } p;
    p.u[0] = pk2(a, b); p.u[1] = pk2(c, d);
    return p.q;
}

__global__ __launch_bounds__(256)
void qlinear_fused_wmma(const float* __restrict__ X, const float* __restrict__ W,
                        const float* __restrict__ alpha, const float* __restrict__ bias,
                        float* __restrict__ Y, int M, int N, int K)
{
    __shared__ __align__(16) _Float16 As[2][FBM * LDT];
    __shared__ __align__(16) _Float16 Bs[2][FBN * LDT];

    const int tid   = threadIdx.x;
    const int lane  = tid & 31;
    const int wave  = tid >> 5;
    const int waveM = wave & 3;
    const int waveN = wave >> 2;
    const int blockM = blockIdx.y * FBM;
    const int blockN = blockIdx.x * FBN;

    const float alpha_eff = __builtin_fabsf(alpha[0]) + 1e-8f;
    const float qth       = 0.5f * alpha_eff;

    const int lr = tid >> 3;
    const int lc = (tid & 7) << 2;
    const float* xa = X + (size_t)(blockM + lr) * K + lc;
    const float* wa = W + (size_t)(blockN + lr) * K + lc;

    float4 aReg[4], bReg[4];
    const int KT = K / BK;

    auto loadRegs = [&](int kt) {
        const int k0 = kt * BK;
#pragma unroll
        for (int p = 0; p < 4; ++p) {
            aReg[p] = *(const float4*)(xa + (size_t)(32 * p) * K + k0);
            bReg[p] = *(const float4*)(wa + (size_t)(32 * p) * K + k0);
        }
    };
    auto storeLDS = [&](int st) {
#pragma unroll
        for (int p = 0; p < 4; ++p) {
            const int idx = (lr + 32 * p) * LDT + lc;
            *(unsigned long long*)&As[st][idx] =
                pack4h(aReg[p].x, aReg[p].y, aReg[p].z, aReg[p].w);
            *(unsigned long long*)&Bs[st][idx] =
                pack4h(qtern(bReg[p].x, qth), qtern(bReg[p].y, qth),
                       qtern(bReg[p].z, qth), qtern(bReg[p].w, qth));
        }
    };

    v8f acc[2][4];
#pragma unroll
    for (int i = 0; i < 2; ++i)
#pragma unroll
        for (int j = 0; j < 4; ++j)
            acc[i][j] = (v8f)(0.0f);

    loadRegs(0); storeLDS(0);
    __syncthreads();

    for (int kt = 0; kt < KT; ++kt) {
        const int cur = kt & 1;
        const bool haveNext = (kt + 1) < KT;
        if (haveNext) loadRegs(kt + 1);

        const _Float16* Ab = &As[cur][(waveM * 32) * LDT];
        const _Float16* Bb = &Bs[cur][(waveN * 64) * LDT];
        v16h af0 = load_frag(Ab, lane);
        v16h af1 = load_frag(Ab + 16 * LDT, lane);
        v16h bf[4];
#pragma unroll
        for (int j = 0; j < 4; ++j)
            bf[j] = load_frag(Bb + j * 16 * LDT, lane);
#pragma unroll
        for (int j = 0; j < 4; ++j) {
            acc[0][j] = __builtin_amdgcn_wmma_f32_16x16x32_f16(
                false, af0, false, bf[j], (short)0, acc[0][j], false, false);
            acc[1][j] = __builtin_amdgcn_wmma_f32_16x16x32_f16(
                false, af1, false, bf[j], (short)0, acc[1][j], false, false);
        }
        if (haveNext) storeLDS((kt + 1) & 1);
        __syncthreads();
    }

#pragma unroll
    for (int i = 0; i < 2; ++i) {
        const int rbase = blockM + waveM * 32 + i * 16 + ((lane & 16) ? 8 : 0);
#pragma unroll
        for (int j = 0; j < 4; ++j) {
            const int col = blockN + waveN * 64 + j * 16 + (lane & 15);
            const float bv = bias[col];
#pragma unroll
            for (int v = 0; v < 8; ++v)
                Y[(size_t)(rbase + v) * N + col] = alpha_eff * acc[i][j][v] + bv;
        }
    }
}

// ---------- launch ----------

extern "C" void kernel_launch(void* const* d_in, const int* in_sizes, int n_in,
                              void* d_out, int out_size, void* d_ws, size_t ws_size,
                              hipStream_t stream) {
    const float* x     = (const float*)d_in[0];
    const float* w     = (const float*)d_in[1];
    const float* alpha = (const float*)d_in[2];
    const float* bias  = (const float*)d_in[3];
    float*       y     = (float*)d_out;

    const int OUT = in_sizes[3];            // 4096
    const int K   = in_sizes[1] / OUT;      // 4096
    const int M   = in_sizes[0] / K;        // 8192

    const size_t needX = (size_t)M * K * sizeof(_Float16);
    const size_t needQ = (size_t)OUT * K * sizeof(_Float16);

    if (ws_size >= needX + needQ) {
        _Float16* Xh = (_Float16*)d_ws;
        _Float16* Qh = (_Float16*)((char*)d_ws + needX);

        const long n8x = (long)M * K / 8;
        const long n8w = (long)OUT * K / 8;
        pack_x_f16<<<(unsigned)((n8x + 255) / 256), 256, 0, stream>>>(x, (u32x4*)Xh, n8x);
        quant_w_f16<<<(unsigned)((n8w + 255) / 256), 256, 0, stream>>>(w, (u32x4*)Qh, alpha, n8w);

        dim3 grid(OUT / GBN, M / GBM);      // 16 x 64 = 1024 workgroups
        gemm_f16_wmma<<<grid, 256, 0, stream>>>(Xh, Qh, alpha, bias, y, M, OUT, K);
    } else {
        dim3 grid(OUT / FBN, M / FBM);      // 32 x 64 = 2048 workgroups
        qlinear_fused_wmma<<<grid, 256, 0, stream>>>(x, w, alpha, bias, y, M, OUT, K);
    }
}
